// MultiHeadAttentionSpatial_3728031613483
// MI455X (gfx1250) — compile-verified, matrix operand order checked by reading, not run on hardware
//
#include <hip/hip_runtime.h>
#include <hip/hip_bf16.h>
#include <math.h>

typedef __attribute__((ext_vector_type(16))) _Float16 v16h;
typedef __attribute__((ext_vector_type(8)))  float    v8f;

#define D_MODEL 768
#define N_HEAD  12
#define D_HEAD  64
#define SEQ     1024
#define BATCH   4
#define MROWS   (BATCH*SEQ)   /* 4096 */
#define SD      5
#define LN_EPS  1e-5f

// k-index mapping inside a 16-bit 16x32 A/B fragment (wave32, ISA 7.12.2):
// element pair j (j=0..7) of lane L: K base = (j<4 ? ko+2j : 16+ko+2(j-4)), ko = (L&16)?8:0
__device__ __forceinline__ int frag_k(int j, int ko) {
    return (j < 4) ? (ko + 2 * j) : (16 + ko + 2 * (j - 4));
}

// ---------------------------------------------------------------- f32 -> f16
__global__ void cvt_f16_kernel(const float* __restrict__ src, _Float16* __restrict__ dst, int n) {
    int i = blockIdx.x * blockDim.x + threadIdx.x;
    if (i < n) dst[i] = (_Float16)src[i];
}

// ------------------------------------------- Y(f16) = X(f16) @ W^T(f16) + b
// one 16x16 tile per wave; W row-major [N][K], so B[k][n]=W[n][k] is a
// contiguous read along k (lowered to global_load_b128 per lane).
__global__ void gemm_nt_h(const _Float16* __restrict__ X, const _Float16* __restrict__ W,
                          const float* __restrict__ bias, _Float16* __restrict__ Y,
                          int M, int N, int K) {
    const int wave = threadIdx.x >> 5;
    const int lane = threadIdx.x & 31;
    const int mt = blockIdx.y * (blockDim.x >> 5) + wave;
    const int nt = blockIdx.x;
    if (mt * 16 >= M) return;                      // wave-uniform
    const int m0 = mt * 16, n0 = nt * 16;
    const int mA = lane & 15;
    const int ko = (lane & 16) ? 8 : 0;
    const _Float16* xrow = X + (size_t)(m0 + mA) * K;
    const _Float16* wrow = W + (size_t)(n0 + mA) * K;
    v8f c = {};
    for (int k0 = 0; k0 < K; k0 += 32) {
        v16h a, b;
#pragma unroll
        for (int j = 0; j < 8; ++j) {
            int kk = frag_k(j, ko);
            a[2*j]   = xrow[k0 + kk];
            a[2*j+1] = xrow[k0 + kk + 1];
            b[2*j]   = wrow[k0 + kk];
            b[2*j+1] = wrow[k0 + kk + 1];
        }
        c = __builtin_amdgcn_wmma_f32_16x16x32_f16(false, a, false, b, (short)0, c, false, false);
    }
    const int mc = (lane & 16) ? 8 : 0;
    const int n  = n0 + (lane & 15);
    const float bn = bias[n];
#pragma unroll
    for (int r = 0; r < 8; ++r)
        Y[(size_t)(m0 + r + mc) * N + n] = (_Float16)(c[r] + bn);
}

// ---------------- V projection with transposed epilogue:
// Yt layout [b][channel(768)][t(1024)] f16, so the P@V kernel can read V^T
// contiguously along t. Tiles never straddle a batch (1024 % 16 == 0).
__global__ void gemm_nt_h_tr(const _Float16* __restrict__ X, const _Float16* __restrict__ W,
                             const float* __restrict__ bias, _Float16* __restrict__ Yt,
                             int M, int N, int K) {
    const int wave = threadIdx.x >> 5;
    const int lane = threadIdx.x & 31;
    const int mt = blockIdx.y * (blockDim.x >> 5) + wave;
    const int nt = blockIdx.x;
    if (mt * 16 >= M) return;
    const int m0 = mt * 16, n0 = nt * 16;
    const int mA = lane & 15;
    const int ko = (lane & 16) ? 8 : 0;
    const _Float16* xrow = X + (size_t)(m0 + mA) * K;
    const _Float16* wrow = W + (size_t)(n0 + mA) * K;
    v8f c = {};
    for (int k0 = 0; k0 < K; k0 += 32) {
        v16h a, b;
#pragma unroll
        for (int j = 0; j < 8; ++j) {
            int kk = frag_k(j, ko);
            a[2*j]   = xrow[k0 + kk];
            a[2*j+1] = xrow[k0 + kk + 1];
            b[2*j]   = wrow[k0 + kk];
            b[2*j+1] = wrow[k0 + kk + 1];
        }
        c = __builtin_amdgcn_wmma_f32_16x16x32_f16(false, a, false, b, (short)0, c, false, false);
    }
    const int mc = (lane & 16) ? 8 : 0;
    const int n  = n0 + (lane & 15);
    const float bn = bias[n];
    const int b_  = m0 / SEQ;                      // whole tile in one batch
    const int t0_ = m0 - b_ * SEQ;
    _Float16* ycol = Yt + ((size_t)b_ * D_MODEL + n) * SEQ + t0_ + mc;
#pragma unroll
    for (int r = 0; r < 8; ++r)
        ycol[r] = (_Float16)(c[r] + bn);           // 8 consecutive t per lane
}

// ------------------------------------------- Y(f32) = X(f16) @ W^T(f16) + b
__global__ void gemm_nt_s(const _Float16* __restrict__ X, const _Float16* __restrict__ W,
                          const float* __restrict__ bias, float* __restrict__ Y,
                          int M, int N, int K) {
    const int wave = threadIdx.x >> 5;
    const int lane = threadIdx.x & 31;
    const int mt = blockIdx.y * (blockDim.x >> 5) + wave;
    const int nt = blockIdx.x;
    if (mt * 16 >= M) return;
    const int m0 = mt * 16, n0 = nt * 16;
    const int mA = lane & 15;
    const int ko = (lane & 16) ? 8 : 0;
    const _Float16* xrow = X + (size_t)(m0 + mA) * K;
    const _Float16* wrow = W + (size_t)(n0 + mA) * K;
    v8f c = {};
    for (int k0 = 0; k0 < K; k0 += 32) {
        v16h a, b;
#pragma unroll
        for (int j = 0; j < 8; ++j) {
            int kk = frag_k(j, ko);
            a[2*j]   = xrow[k0 + kk];
            a[2*j+1] = xrow[k0 + kk + 1];
            b[2*j]   = wrow[k0 + kk];
            b[2*j+1] = wrow[k0 + kk + 1];
        }
        c = __builtin_amdgcn_wmma_f32_16x16x32_f16(false, a, false, b, (short)0, c, false, false);
    }
    const int mc = (lane & 16) ? 8 : 0;
    const int n  = n0 + (lane & 15);
    const float bn = bias[n];
#pragma unroll
    for (int r = 0; r < 8; ++r)
        Y[(size_t)(m0 + r + mc) * N + n] = c[r] + bn;
}

// ---------------- fused scores: logits = log(clip(relu(loc),1e-6)) + QK^T*s
__global__ void attn_scores(const _Float16* __restrict__ qp, const _Float16* __restrict__ kp,
                            const float* __restrict__ plocs, const unsigned char* __restrict__ kpm,
                            const float* __restrict__ wloc, const float* __restrict__ bloc,
                            float* __restrict__ fused) {
    const int wave  = threadIdx.x >> 5;
    const int lane  = threadIdx.x & 31;
    const int ltile = blockIdx.x * (blockDim.x >> 5) + wave;   // 0..63
    const int h = blockIdx.y >> 2;
    const int b = blockIdx.y & 3;
    const int l0 = ltile * 16;
    const int mA = lane & 15;
    const int ko = (lane & 16) ? 8 : 0;
    const float scale = 0.125f;                    // 1/sqrt(64)

    float wl[SD];
#pragma unroll
    for (int s = 0; s < SD; ++s) wl[s] = wloc[h * SD + s];
    const float bl = bloc[h];

    const _Float16* qrow = qp + ((size_t)b * SEQ + l0 + mA) * D_MODEL + h * D_HEAD;
    v16h a0, a1;
#pragma unroll
    for (int j = 0; j < 8; ++j) {
        int kk = frag_k(j, ko);
        a0[2*j]   = qrow[kk];        a0[2*j+1] = qrow[kk + 1];
        a1[2*j]   = qrow[32 + kk];   a1[2*j+1] = qrow[32 + kk + 1];
    }

    float* frow = fused + ((size_t)(h * BATCH + b) * SEQ + l0) * SEQ;

    for (int t0 = 0; t0 < SEQ; t0 += 16) {
        const _Float16* krow = kp + ((size_t)b * SEQ + t0 + mA) * D_MODEL + h * D_HEAD;
        v16h b0, b1;
#pragma unroll
        for (int j = 0; j < 8; ++j) {
            int kk = frag_k(j, ko);
            b0[2*j]   = krow[kk];        b0[2*j+1] = krow[kk + 1];
            b1[2*j]   = krow[32 + kk];   b1[2*j+1] = krow[32 + kk + 1];
        }
        v8f c = {};
        c = __builtin_amdgcn_wmma_f32_16x16x32_f16(false, a0, false, b0, (short)0, c, false, false);
        c = __builtin_amdgcn_wmma_f32_16x16x32_f16(false, a1, false, b1, (short)0, c, false, false);

        const int t = t0 + (lane & 15);
        const bool masked = kpm[b * SEQ + t] != 0;
        const int mc = (lane & 16) ? 8 : 0;

        // prefetch next tile's pairwise_locs rows (global_prefetch_b8)
        if (t0 + 16 < SEQ) {
            const float* pnext = plocs + (((size_t)b * SEQ + l0 + mc) * SEQ + (t + 16)) * SD;
            __builtin_prefetch(pnext, 0, 1);
        }
#pragma unroll
        for (int r = 0; r < 8; ++r) {
            const int l = l0 + r + mc;
            float logit;
            if (masked) {
                logit = -INFINITY;
            } else {
                const float* pl = plocs + (((size_t)b * SEQ + l) * SEQ + t) * SD;
                float loc = bl;
#pragma unroll
                for (int s = 0; s < SD; ++s) loc = fmaf(pl[s], wl[s], loc);
                loc = fmaxf(loc, 0.0f);
                logit = __logf(fmaxf(loc, 1e-6f)) + c[r] * scale;
            }
            frow[(size_t)(r + mc) * SEQ + t] = logit;
        }
    }
}

// -------------------------- in-place row softmax over 1024, register-staged
__global__ void softmax_rows(float* __restrict__ data) {
    __shared__ float red[256];
    float* row = data + (size_t)blockIdx.x * SEQ;
    const int tid = threadIdx.x;
    float x[4];
    float lmax = -INFINITY;
#pragma unroll
    for (int i = 0; i < 4; ++i) { x[i] = row[tid + i * 256]; lmax = fmaxf(lmax, x[i]); }
    red[tid] = lmax; __syncthreads();
    for (int s = 128; s > 0; s >>= 1) {
        if (tid < s) red[tid] = fmaxf(red[tid], red[tid + s]);
        __syncthreads();
    }
    const float m = red[0]; __syncthreads();
    float lsum = 0.0f;
#pragma unroll
    for (int i = 0; i < 4; ++i) { x[i] = __expf(x[i] - m); lsum += x[i]; }
    red[tid] = lsum; __syncthreads();
    for (int s = 128; s > 0; s >>= 1) {
        if (tid < s) red[tid] += red[tid + s];
        __syncthreads();
    }
    const float inv = 1.0f / red[0];
#pragma unroll
    for (int i = 0; i < 4; ++i) row[tid + i * 256] = x[i] * inv;
}

// ------------- out_h = P(f32->f16) @ V_h ; V supplied transposed (vt16) so
// B-fragment reads are contiguous 16B chunks along t.
__global__ void attn_v(const float* __restrict__ fused, const _Float16* __restrict__ vt,
                       _Float16* __restrict__ ao) {
    const int wave  = threadIdx.x >> 5;
    const int lane  = threadIdx.x & 31;
    const int ltile = blockIdx.x * (blockDim.x >> 5) + wave;
    const int h = blockIdx.y >> 2;
    const int b = blockIdx.y & 3;
    const int l0 = ltile * 16;
    const int mA = lane & 15;
    const int ko = (lane & 16) ? 8 : 0;
    const int nL = lane & 15;

    const float* prow = fused + ((size_t)(h * BATCH + b) * SEQ + l0 + mA) * SEQ;
    // vt row for this lane's output column, per n-subtile
    const _Float16* vrow0 = vt + ((size_t)b * D_MODEL + h * D_HEAD + nL) * SEQ;

    v8f acc0 = {}, acc1 = {}, acc2 = {}, acc3 = {};
    for (int k0 = 0; k0 < SEQ; k0 += 32) {
        v16h a;
#pragma unroll
        for (int j = 0; j < 8; ++j) {
            int kk = frag_k(j, ko);
            a[2*j]   = (_Float16)prow[k0 + kk];
            a[2*j+1] = (_Float16)prow[k0 + kk + 1];
        }
#pragma unroll
        for (int nsub = 0; nsub < 4; ++nsub) {
            const _Float16* vrow = vrow0 + (size_t)(nsub * 16) * SEQ;
            v16h bb;
#pragma unroll
            for (int j = 0; j < 8; ++j) {
                int kk = frag_k(j, ko);
                bb[2*j]   = vrow[k0 + kk];         // contiguous along t
                bb[2*j+1] = vrow[k0 + kk + 1];
            }
            v8f* accp = (nsub == 0) ? &acc0 : (nsub == 1) ? &acc1 : (nsub == 2) ? &acc2 : &acc3;
            *accp = __builtin_amdgcn_wmma_f32_16x16x32_f16(false, a, false, bb, (short)0, *accp, false, false);
        }
    }
    const int mc = (lane & 16) ? 8 : 0;
    v8f accs[4] = { acc0, acc1, acc2, acc3 };
#pragma unroll
    for (int nsub = 0; nsub < 4; ++nsub) {
        const int n = h * D_HEAD + nsub * 16 + nL;
#pragma unroll
        for (int r = 0; r < 8; ++r)
            ao[((size_t)b * SEQ + l0 + r + mc) * D_MODEL + n] = (_Float16)accs[nsub][r];
    }
}

// -------------------------------------------- residual add + LayerNorm(768)
__global__ void fc_ln(const float* __restrict__ fcout, const float* __restrict__ resid,
                      const float* __restrict__ g, const float* __restrict__ bta,
                      float* __restrict__ out) {
    __shared__ float red[256];
    const int row = blockIdx.x;
    const int tid = threadIdx.x;
    const float* fr = fcout + (size_t)row * D_MODEL;
    const float* rr = resid + (size_t)row * D_MODEL;
    float x[3];
    float s = 0.0f, s2 = 0.0f;
#pragma unroll
    for (int i = 0; i < 3; ++i) {
        int c = tid + i * 256;
        x[i] = fr[c] + rr[c];
        s += x[i]; s2 += x[i] * x[i];
    }
    red[tid] = s; __syncthreads();
    for (int st = 128; st > 0; st >>= 1) { if (tid < st) red[tid] += red[tid + st]; __syncthreads(); }
    const float mu = red[0] / (float)D_MODEL; __syncthreads();
    red[tid] = s2; __syncthreads();
    for (int st = 128; st > 0; st >>= 1) { if (tid < st) red[tid] += red[tid + st]; __syncthreads(); }
    const float var = red[0] / (float)D_MODEL - mu * mu;
    const float inv = rsqrtf(var + LN_EPS);
#pragma unroll
    for (int i = 0; i < 3; ++i) {
        int c = tid + i * 256;
        out[(size_t)row * D_MODEL + c] = (x[i] - mu) * inv * g[c] + bta[c];
    }
}

extern "C" void kernel_launch(void* const* d_in, const int* in_sizes, int n_in,
                              void* d_out, int out_size, void* d_ws, size_t ws_size,
                              hipStream_t stream) {
    (void)in_sizes; (void)n_in; (void)out_size; (void)ws_size;

    const float* q     = (const float*)d_in[0];
    const float* k     = (const float*)d_in[1];
    const float* v     = (const float*)d_in[2];
    const float* plocs = (const float*)d_in[3];
    const unsigned char* kpm = (const unsigned char*)d_in[4];   // bool mask, 1 byte
    const float* wq  = (const float*)d_in[5];
    const float* bq  = (const float*)d_in[6];
    const float* wk  = (const float*)d_in[7];
    const float* bk  = (const float*)d_in[8];
    const float* wv  = (const float*)d_in[9];
    const float* bv  = (const float*)d_in[10];
    const float* wfc = (const float*)d_in[11];
    const float* bfc = (const float*)d_in[12];
    const float* wloc = (const float*)d_in[13];
    const float* bloc = (const float*)d_in[14];
    const float* ln_g = (const float*)d_in[15];
    const float* ln_b = (const float*)d_in[16];

    float* out0  = (float*)d_out;
    float* fused = out0 + (size_t)BATCH * SEQ * D_MODEL;        // output #2 region

    // workspace carve-up (~61 MB)
    char*  ws  = (char*)d_ws;
    size_t off = 0;
    auto alloc = [&](size_t bytes) {
        char* p = ws + off;
        off += (bytes + 255) & ~(size_t)255;
        return p;
    };
    const size_t ACT_H = (size_t)MROWS * D_MODEL * sizeof(_Float16);
    const size_t W_H   = (size_t)D_MODEL * D_MODEL * sizeof(_Float16);
    _Float16* q16   = (_Float16*)alloc(ACT_H);
    _Float16* k16   = (_Float16*)alloc(ACT_H);
    _Float16* v16   = (_Float16*)alloc(ACT_H);
    _Float16* wq16  = (_Float16*)alloc(W_H);
    _Float16* wk16  = (_Float16*)alloc(W_H);
    _Float16* wv16  = (_Float16*)alloc(W_H);
    _Float16* wfc16 = (_Float16*)alloc(W_H);
    _Float16* qp16  = (_Float16*)alloc(ACT_H);
    _Float16* kp16  = (_Float16*)alloc(ACT_H);
    _Float16* vt16  = (_Float16*)alloc(ACT_H);   // V^T per batch: [b][768][1024]
    _Float16* ao16  = (_Float16*)alloc(ACT_H);
    float*    fc32  = (float*)   alloc((size_t)MROWS * D_MODEL * sizeof(float));

    // 1) convert activations + weights to f16
    const int nact = MROWS * D_MODEL;
    const int nwt  = D_MODEL * D_MODEL;
    cvt_f16_kernel<<<dim3((nact + 255) / 256), dim3(256), 0, stream>>>(q,   q16,   nact);
    cvt_f16_kernel<<<dim3((nact + 255) / 256), dim3(256), 0, stream>>>(k,   k16,   nact);
    cvt_f16_kernel<<<dim3((nact + 255) / 256), dim3(256), 0, stream>>>(v,   v16,   nact);
    cvt_f16_kernel<<<dim3((nwt  + 255) / 256), dim3(256), 0, stream>>>(wq,  wq16,  nwt);
    cvt_f16_kernel<<<dim3((nwt  + 255) / 256), dim3(256), 0, stream>>>(wk,  wk16,  nwt);
    cvt_f16_kernel<<<dim3((nwt  + 255) / 256), dim3(256), 0, stream>>>(wv,  wv16,  nwt);
    cvt_f16_kernel<<<dim3((nwt  + 255) / 256), dim3(256), 0, stream>>>(wfc, wfc16, nwt);

    // 2) QKV projections (WMMA); V written transposed for the P@V pass
    dim3 gemmGrid(D_MODEL / 16, MROWS / 128);   // (48, 32), 8 waves/block
    gemm_nt_h   <<<gemmGrid, 256, 0, stream>>>(q16, wq16, bq, qp16, MROWS, D_MODEL, D_MODEL);
    gemm_nt_h   <<<gemmGrid, 256, 0, stream>>>(k16, wk16, bk, kp16, MROWS, D_MODEL, D_MODEL);
    gemm_nt_h_tr<<<gemmGrid, 256, 0, stream>>>(v16, wv16, bv, vt16, MROWS, D_MODEL, D_MODEL);

    // 3) fused QK^T + spatial-loc logits (WMMA) -> fused_attn region
    dim3 attnGrid(8, N_HEAD * BATCH);           // 64 l-tiles x 48 (h,b)
    attn_scores<<<attnGrid, 256, 0, stream>>>(qp16, kp16, plocs, kpm, wloc, bloc, fused);

    // 4) softmax rows (in place -> final probabilities output)
    softmax_rows<<<dim3(N_HEAD * BATCH * SEQ), 256, 0, stream>>>(fused);

    // 5) P @ V (WMMA), merged head-major -> [b][l][768] f16
    attn_v<<<attnGrid, 256, 0, stream>>>(fused, vt16, ao16);

    // 6) FC projection (WMMA, f32 out)
    gemm_nt_s<<<gemmGrid, 256, 0, stream>>>(ao16, wfc16, bfc, fc32, MROWS, D_MODEL, D_MODEL);

    // 7) residual + LayerNorm -> output #1
    fc_ln<<<dim3(MROWS), 256, 0, stream>>>(fc32, q, ln_g, ln_b, out0);
}
